// GroupedMLP_53850299957847
// MI455X (gfx1250) — compile-verified, matrix operand order checked by reading, not run on hardware
//
#include <hip/hip_runtime.h>

#define NE  8
#define HID 1024
#define FFN 4096
#define TOK 16384

typedef __attribute__((ext_vector_type(16))) __bf16 v16bf;
typedef __attribute__((ext_vector_type(8)))  __bf16 v8bf;
typedef __attribute__((ext_vector_type(8)))  float  v8f_t;
typedef __attribute__((ext_vector_type(4)))  int    v4i;
typedef __attribute__((ext_vector_type(4)))  unsigned v4u;
typedef __attribute__((ext_vector_type(8)))  int    v8i;

#define AS_GLOBAL __attribute__((address_space(1)))
#define AS_LDS    __attribute__((address_space(3)))

#if __has_builtin(__builtin_amdgcn_tensor_load_to_lds)
#define USE_TDM 1
#else
#define USE_TDM 0
#endif

struct alignas(16) BF8 { __bf16 v[8]; };

// ---------------------------------------------------------------------------
// CDNA5 async global->LDS 16B copy (fallback staging path)
// ---------------------------------------------------------------------------
__device__ __forceinline__ void async_copy16(const __bf16* g, __bf16* l) {
#if __has_builtin(__builtin_amdgcn_global_load_async_to_lds_b128)
  AS_GLOBAL v4i* gp = (AS_GLOBAL v4i*)(uintptr_t)g;
  AS_LDS    v4i* lp = (AS_LDS    v4i*)(unsigned)(uintptr_t)l;
  __builtin_amdgcn_global_load_async_to_lds_b128(gp, lp, 0, 0);
#else
  unsigned laddr = (unsigned)(uintptr_t)l;
  asm volatile("global_load_async_to_lds_b128 %0, %1, off"
               :: "v"(laddr), "v"(g) : "memory");
#endif
}

__device__ __forceinline__ void wait_async0() {
#if __has_builtin(__builtin_amdgcn_s_wait_asynccnt)
  __builtin_amdgcn_s_wait_asynccnt(0);
#else
  asm volatile("s_wait_asynccnt 0x0" ::: "memory");
#endif
}

__device__ __forceinline__ void stage_tile_async(const __bf16* __restrict__ src, int ldK,
                                                 __bf16* __restrict__ dst, int tid) {
#pragma unroll
  for (int i = 0; i < 2; ++i) {
    const int chunk = tid + i * 256;       // 0..511
    const int row   = chunk >> 2;          // 0..127
    const int kc    = (chunk & 3) << 3;    // 0,8,16,24
    async_copy16(src + (size_t)row * ldK + kc, dst + row * 32 + kc);
  }
}

#if USE_TDM
// ---------------------------------------------------------------------------
// Tensor Data Mover: one-instruction DMA of a 128-row x 32-elem bf16 tile
// (row = 64 B contiguous, row stride = rowlen*2 B) into LDS, rows packed.
// D# per CDNA5 ISA ch.8: group0 {count, lds_addr, global_addr, type=2},
// group1 {mask=0, data_size=2B, tensor_dim0=rowlen, tensor_dim1=128,
//         tile_dim0=32, tile_dim1=128, dim0_stride=rowlen}, groups 2/3 = 0.
// Builtin arity on this toolchain: 6 args (g0, g1, g2, g3, extra8, cpol).
// ---------------------------------------------------------------------------
__device__ __forceinline__ void tdm_stage_tile(const __bf16* gsrc, unsigned lds_addr,
                                               int rowlen) {
  const unsigned long long ga = (unsigned long long)(uintptr_t)gsrc;
  v4u g0;
  g0.x = 1u;                                                 // count=1, no gather
  g0.y = lds_addr;                                           // LDS byte address
  g0.z = (unsigned)ga;                                       // global_addr[31:0]
  g0.w = (unsigned)((ga >> 32) & 0x01FFFFFFu) | (2u << 30);  // ga[56:32] | type=2
  v8i g1;
  g1[0] = 1 << 16;                                   // wg_mask=0, data_size=1 (2 B)
  g1[1] = (rowlen & 0xFFFF) << 16;                   // tensor_dim0[15:0]
  g1[2] = ((rowlen >> 16) & 0xFFFF) | (128 << 16);   // tensor_dim0[31:16] | tensor_dim1=128
  g1[3] = 32 << 16;                                  // tensor_dim1 hi=0 | tile_dim0=32
  g1[4] = 128;                                       // tile_dim1=128 | tile_dim2=0
  g1[5] = rowlen;                                    // tensor_dim0_stride[31:0]
  g1[6] = 0;                                         // stride hi / dim1_stride
  g1[7] = 0;
  const v4i z4 = {0, 0, 0, 0};                       // groups 2/3 unused (2D tile)
  const v8i z8 = {0, 0, 0, 0, 0, 0, 0, 0};           // extra descriptor words unused
  __builtin_amdgcn_tensor_load_to_lds(g0, g1, z4, z4, z8, 0);
}
#endif

// ---------------------------------------------------------------------------
// f32 -> bf16 elementwise convert, 8 elements/thread
// ---------------------------------------------------------------------------
__global__ __launch_bounds__(256)
void k_convert(const float* __restrict__ src, __bf16* __restrict__ dst) {
  const size_t i = (size_t)blockIdx.x * 256 + threadIdx.x;
  const float4 f0 = ((const float4*)src)[i * 2 + 0];
  const float4 f1 = ((const float4*)src)[i * 2 + 1];
  BF8 r;
  r.v[0] = (__bf16)f0.x; r.v[1] = (__bf16)f0.y;
  r.v[2] = (__bf16)f0.z; r.v[3] = (__bf16)f0.w;
  r.v[4] = (__bf16)f1.x; r.v[5] = (__bf16)f1.y;
  r.v[6] = (__bf16)f1.z; r.v[7] = (__bf16)f1.w;
  *(BF8*)(dst + i * 8) = r;
}

// ---------------------------------------------------------------------------
// w2 [NE][FFN][HID] f32  ->  w2t [NE][HID][FFN] bf16   (LDS-tiled transpose)
// ---------------------------------------------------------------------------
__global__ __launch_bounds__(256)
void k_transpose_w2(const float* __restrict__ w2, __bf16* __restrict__ w2t) {
  __shared__ float t[32][33];
  const int tx = threadIdx.x, ty = threadIdx.y;
  const int f0 = blockIdx.x * 32;
  const int n0 = blockIdx.y * 32;
  const int g  = blockIdx.z;
  const float* src = w2 + (size_t)g * FFN * HID;
  __bf16*      dst = w2t + (size_t)g * HID * FFN;
#pragma unroll
  for (int j = 0; j < 32; j += 8)
    t[ty + j][tx] = src[(size_t)(f0 + ty + j) * HID + (n0 + tx)];
  __syncthreads();
#pragma unroll
  for (int j = 0; j < 32; j += 8)
    dst[(size_t)(n0 + ty + j) * FFN + (f0 + tx)] = (__bf16)t[tx][ty + j];
}

// ISA 16-bit A/B fragment layout: lane(0..15) holds K 0..7 & 16..23,
// lane(16..31) holds K 8..15 & 24..31 -> two 16B chunks at half*8 and 16+half*8.
__device__ __forceinline__ v16bf frag_from_lds(const __bf16* p) {
  v8bf lo = *(const v8bf*)p;
  v8bf hi = *(const v8bf*)(p + 16);
  return __builtin_shufflevector(lo, hi, 0, 1, 2, 3, 4, 5, 6, 7,
                                        8, 9, 10, 11, 12, 13, 14, 15);
}

// ---------------------------------------------------------------------------
// NT grouped GEMM:  C[M,N] = A[M,K] * B_g[N,K]^T   (expert g from row prefix)
// block tile 128x128, 8 wave32s (2x4), wave tile 64x32, K step 32,
// double-buffered LDS staged via the Tensor Data Mover (TENSORcnt-tracked,
// one instruction per tile, issued by waves 0/1), v_wmma_16x16x32_bf16 core.
// ---------------------------------------------------------------------------
template <bool GELU, typename OutT>
__global__ __launch_bounds__(256)
void k_gemm_nt(const __bf16* __restrict__ A, const __bf16* __restrict__ B,
               const int* __restrict__ tpe, OutT* __restrict__ Out,
               int N, int K) {
  __shared__ __attribute__((aligned(16))) __bf16 ldsA[2][128 * 32];
  __shared__ __attribute__((aligned(16))) __bf16 ldsB[2][128 * 32];

  const int tid  = threadIdx.x;
  const int lane = tid & 31;
  const int wm   = (tid >> 5) >> 2;  // 0..1 : wave row   (64 rows)
  const int wn   = (tid >> 5) & 3;   // 0..3 : wave col   (32 cols)
  const int half = lane >> 4;        // 0..1
  const int lr   = lane & 15;        // 0..15

  const int n0 = blockIdx.x * 128;
  const int m0 = blockIdx.y * 128;

  // expert for this M tile (tile boundaries align with per-expert counts)
  int g = 0, pre = 0;
#pragma unroll
  for (int e = 0; e < NE - 1; ++e) {
    pre += tpe[e];
    if (m0 >= pre) g = e + 1;
  }

  const __bf16* Ab = A + (size_t)m0 * K;
  const __bf16* Bb = B + (size_t)g * N * K + (size_t)n0 * K;

#if USE_TDM
  const unsigned ldsAaddr = (unsigned)(uintptr_t)&ldsA[0][0];
  const unsigned ldsBaddr = (unsigned)(uintptr_t)&ldsB[0][0];
#endif

  v8f_t acc[4][2] = {};

  // stage tile 0 into buffer 0
#if USE_TDM
  if (tid < 32)       tdm_stage_tile(Ab, ldsAaddr, K);
  else if (tid < 64)  tdm_stage_tile(Bb, ldsBaddr, K);
#else
  stage_tile_async(Ab, K, ldsA[0], tid);
  stage_tile_async(Bb, K, ldsB[0], tid);
#endif

  const int KT = K >> 5;
  for (int kt = 0; kt < KT; ++kt) {
#if USE_TDM
    if (tid < 64) __builtin_amdgcn_s_wait_tensorcnt(0);  // DMA waves drain TENSORcnt
#else
    wait_async0();
#endif
    __syncthreads();     // staged tile visible to all waves; prior reads drained
    const int cur = kt & 1;
    if (kt + 1 < KT) {   // prefetch next tile into the spare buffer
#if USE_TDM
      if (tid < 32)      tdm_stage_tile(Ab + (size_t)(kt + 1) * 32,
                                        ldsAaddr + (unsigned)(cur ^ 1) * 8192u, K);
      else if (tid < 64) tdm_stage_tile(Bb + (size_t)(kt + 1) * 32,
                                        ldsBaddr + (unsigned)(cur ^ 1) * 8192u, K);
#else
      stage_tile_async(Ab + (size_t)(kt + 1) * 32, K, ldsA[cur ^ 1], tid);
      stage_tile_async(Bb + (size_t)(kt + 1) * 32, K, ldsB[cur ^ 1], tid);
#endif
    }

    v16bf af[4], bfr[2];
#pragma unroll
    for (int i = 0; i < 4; ++i)
      af[i] = frag_from_lds(&ldsA[cur][(wm * 64 + i * 16 + lr) * 32 + half * 8]);
#pragma unroll
    for (int j = 0; j < 2; ++j)
      bfr[j] = frag_from_lds(&ldsB[cur][(wn * 32 + j * 16 + lr) * 32 + half * 8]);

#pragma unroll
    for (int i = 0; i < 4; ++i)
#pragma unroll
      for (int j = 0; j < 2; ++j)
        acc[i][j] = __builtin_amdgcn_wmma_f32_16x16x32_bf16(
            false, af[i], false, bfr[j], (short)0, acc[i][j], false, false);
  }

  // C/D layout: VGPR r -> M = r (+8 for lanes 16..31), lane%16 -> N
#pragma unroll
  for (int i = 0; i < 4; ++i)
#pragma unroll
    for (int j = 0; j < 2; ++j) {
      const int mb = m0 + wm * 64 + i * 16 + half * 8;
      const int nn = n0 + wn * 32 + j * 16 + lr;
#pragma unroll
      for (int r = 0; r < 8; ++r) {
        float v = acc[i][j][r];
        if constexpr (GELU)
          v = 0.5f * v * (1.0f + erff(v * 0.70710678118654752f));
        Out[(size_t)(mb + r) * N + nn] = (OutT)v;
      }
    }
}

// ---------------------------------------------------------------------------
extern "C" void kernel_launch(void* const* d_in, const int* in_sizes, int n_in,
                              void* d_out, int out_size, void* d_ws, size_t ws_size,
                              hipStream_t stream) {
  (void)in_sizes; (void)n_in; (void)out_size; (void)ws_size;
  const float* x   = (const float*)d_in[0];
  const int*   tpe = (const int*)d_in[1];
  const float* w1  = (const float*)d_in[2];
  const float* w2  = (const float*)d_in[3];
  float* out = (float*)d_out;

  char* ws = (char*)d_ws;
  __bf16* xb  = (__bf16*)ws;                                      //  32 MiB
  __bf16* w1b = (__bf16*)(ws + (size_t)TOK * HID * 2);            //  64 MiB
  __bf16* w2t = (__bf16*)(ws + (size_t)TOK * HID * 2
                             + (size_t)NE * FFN * HID * 2);       //  64 MiB
  __bf16* hb  = (__bf16*)(ws + (size_t)TOK * HID * 2
                             + (size_t)NE * FFN * HID * 2 * 2);   // 128 MiB

  // 0) one-time fp32 -> bf16 conversion of GEMM operands (L2-friendly footprint)
  k_convert<<<(TOK * HID) / (8 * 256), 256, 0, stream>>>(x, xb);
  k_convert<<<(NE * FFN * HID) / (8 * 256), 256, 0, stream>>>(w1, w1b);

  // 1) transpose+convert w2 -> w2t (K-contiguous B for GEMM2)
  k_transpose_w2<<<dim3(FFN / 32, HID / 32, NE), dim3(32, 8), 0, stream>>>(w2, w2t);

  // 2) h = gelu(x @ w1^T)  : M=TOK, N=FFN, K=HID
  k_gemm_nt<true, __bf16>
      <<<dim3(FFN / 128, TOK / 128), 256, 0, stream>>>(xb, w1b, tpe, hb, FFN, HID);

  // 3) out = h @ w2        : M=TOK, N=HID, K=FFN
  k_gemm_nt<false, float>
      <<<dim3(HID / 128, TOK / 128), 256, 0, stream>>>(hb, w2t, tpe, out, HID, FFN);
}
